// PointNetSimple_67748814127145
// MI455X (gfx1250) — matrix operand order repaired and checked by reading, not compiled, not run on hardware
//
#include <hip/hip_runtime.h>
#include <hip/hip_bf16.h>

typedef __attribute__((ext_vector_type(16))) _Float16 v16h;
typedef __attribute__((ext_vector_type(8)))  _Float16 v8h;
typedef __attribute__((ext_vector_type(8)))  float    v8f;

// ---------------------------------------------------------------------------
// Farthest point sampling. One block (1024 threads) per batch. Points and the
// running min-distance array live in registers (CHUNK points per thread,
// strided ownership p = j*1024 + t). Argmax: wave32 shfl reduce, then 32-entry
// LDS cross-wave reduce. Matches reference: sel[0]=0, 1023 serial steps.
// ---------------------------------------------------------------------------
template<int CHUNK>
__global__ void __launch_bounds__(1024)
fps_kernel(const float* __restrict__ xyz, int N, int S, int* __restrict__ fidx) {
    const int b = blockIdx.x;
    const int t = threadIdx.x;
    const float* X = xyz + (size_t)b * N * 3;

    float px[CHUNK], py[CHUNK], pz[CHUNK], md[CHUNK];
#pragma unroll
    for (int j = 0; j < CHUNK; ++j) {
        int p = j * 1024 + t;
        if (p < N) {
            px[j] = X[p * 3 + 0]; py[j] = X[p * 3 + 1]; pz[j] = X[p * 3 + 2];
            md[j] = 1e10f;
        } else {
            px[j] = 0.f; py[j] = 0.f; pz[j] = 0.f;
            md[j] = -1e30f;   // never wins argmax
        }
    }

    __shared__ float wv[32];
    __shared__ int   wi[32];
    __shared__ float selc[3];
    __shared__ int   selS;

    if (t == 0) {
        fidx[(size_t)b * S] = 0;
        selc[0] = X[0]; selc[1] = X[1]; selc[2] = X[2];
    }
    __syncthreads();

    const int lane = t & 31;
    const int wid  = t >> 5;

    for (int it = 1; it < S; ++it) {
        const float cx = selc[0], cy = selc[1], cz = selc[2];
        float bv = -1e30f; int bi = 0;
#pragma unroll
        for (int j = 0; j < CHUNK; ++j) {
            float dx = px[j] - cx, dy = py[j] - cy, dz = pz[j] - cz;
            float d = dx * dx + dy * dy + dz * dz;
            md[j] = fminf(md[j], d);
            if (md[j] > bv) { bv = md[j]; bi = j * 1024 + t; }
        }
        // wave32 argmax (tie -> lower index)
        for (int off = 16; off; off >>= 1) {
            float ov = __shfl_xor(bv, off, 32);
            int   oi = __shfl_xor(bi, off, 32);
            if (ov > bv || (ov == bv && oi < bi)) { bv = ov; bi = oi; }
        }
        if (lane == 0) { wv[wid] = bv; wi[wid] = bi; }
        __syncthreads();
        if (wid == 0) {
            float v2 = wv[lane]; int i2 = wi[lane];
            for (int off = 16; off; off >>= 1) {
                float ov = __shfl_xor(v2, off, 32);
                int   oi = __shfl_xor(i2, off, 32);
                if (ov > v2 || (ov == v2 && oi < i2)) { v2 = ov; i2 = oi; }
            }
            if (lane == 0) { selS = i2; fidx[(size_t)b * S + it] = i2; }
        }
        __syncthreads();
        const int sp = selS;
        if ((sp & 1023) == t) {
            int j = sp >> 10;
            if (j < CHUNK) { selc[0] = px[j]; selc[1] = py[j]; selc[2] = pz[j]; }
        }
        __syncthreads();
    }
}

// ---------------------------------------------------------------------------
// Ball query: one wave per query. Scans points in index order 32 at a time;
// ballot + prefix popcount appends the first-K in-radius indices in ascending
// index order (== reference top_k-on-(N-idx) semantics). Pads with first hit.
// Also gathers new_xyz.
// ---------------------------------------------------------------------------
__global__ void ballquery_kernel(const float* __restrict__ xyz, const int* __restrict__ fidx,
                                 int N, int S, int K, float r2,
                                 int* __restrict__ idx, float* __restrict__ nxyz) {
    const int lane = threadIdx.x & 31;
    const int q = blockIdx.x * (blockDim.x >> 5) + (threadIdx.x >> 5);
    const int b = q / S;
    const float* X = xyz + (size_t)b * N * 3;
    const int fi = fidx[q];
    const float qx = X[fi * 3 + 0], qy = X[fi * 3 + 1], qz = X[fi * 3 + 2];
    if (lane == 0) {
        nxyz[(size_t)q * 3 + 0] = qx;
        nxyz[(size_t)q * 3 + 1] = qy;
        nxyz[(size_t)q * 3 + 2] = qz;
    }
    int* out = idx + (size_t)q * K;
    int cnt = 0, first = 0;
    bool haveFirst = false;
    for (int base = 0; base < N; base += 32) {
        const int p = base + lane;
        float dx = X[p * 3 + 0] - qx, dy = X[p * 3 + 1] - qy, dz = X[p * 3 + 2] - qz;
        bool in = (dx * dx + dy * dy + dz * dz) < r2;
        unsigned mask = (unsigned)__ballot(in);
        if (mask) {
            if (!haveFirst) { first = base + __ffs((int)mask) - 1; haveFirst = true; }
            if (in) {
                int pos = cnt + __popc(mask & ((1u << lane) - 1u));
                if (pos < K) out[pos] = p;
            }
            cnt += __popc(mask);
            if (cnt >= K) break;
        }
    }
    if (cnt < K) {
        for (int slot = cnt + lane; slot < K; slot += 32) out[slot] = first;
    }
}

// ---------------------------------------------------------------------------
// Grouping: one thread per (b,s,k) row. Writes padded f16 activation row:
// [xyz_diff(3) | gathered feats(Cf) | zeros up to Cpad].
// ---------------------------------------------------------------------------
__global__ void group_kernel(const float* __restrict__ xyz, const float* __restrict__ feats,
                             const int* __restrict__ idx, const float* __restrict__ nxyz,
                             int N, int S, int K, int Cf, int Cpad,
                             _Float16* __restrict__ Xh) {
    const int m = blockIdx.x * blockDim.x + threadIdx.x;   // total = B*S*K
    const int q = m / K;
    const int b = q / S;
    const int pi = idx[m];
    const float* P = xyz + ((size_t)b * N + pi) * 3;
    const float* Q = nxyz + (size_t)q * 3;
    _Float16* row = Xh + (size_t)m * Cpad;
    row[0] = (_Float16)(P[0] - Q[0]);
    row[1] = (_Float16)(P[1] - Q[1]);
    row[2] = (_Float16)(P[2] - Q[2]);
    int c = 3;
    if (Cf > 0) {
        const float* F = feats + ((size_t)b * N + pi) * Cf;
        for (int i = 0; i < Cf; ++i) row[3 + i] = (_Float16)F[i];
        c = 3 + Cf;
    }
    for (; c < Cpad; ++c) row[c] = (_Float16)0.f;
}

// ---------------------------------------------------------------------------
// Pack weights into WMMA B-fragment layout: Bp[(ntile*nchunks+chunk)*32+lane][e]
// with N = ntile*16 + (lane&15), K = chunk*32 + (lane>=16?8:0) + (e<8?e:e+8). Zero-pad
// K >= Cin.
// ---------------------------------------------------------------------------
__global__ void packw_kernel(const float* __restrict__ W, int Cin, int Cout, int nchunks,
                             _Float16* __restrict__ Bp) {
    const int ntile = blockIdx.x / nchunks;
    const int chunk = blockIdx.x % nchunks;
    const int lane = threadIdx.x;
    _Float16* dst = Bp + ((size_t)blockIdx.x * 32 + lane) * 16;
    const int n = ntile * 16 + (lane & 15);
    const int kbase = chunk * 32 + ((lane >= 16) ? 8 : 0);
#pragma unroll
    for (int e = 0; e < 16; ++e) {
        int k = kbase + (e < 8 ? e : e + 8);
        dst[e] = (_Float16)((k < Cin) ? W[(size_t)k * Cout + n] : 0.f);
    }
}

// ---------------------------------------------------------------------------
// WMMA GEMM: Y(M x Cout) = Xh(M x Cpad, f16 padded) @ W. One wave per 16x16
// output tile; block = 8 waves = 8 consecutive M-tiles at one N-tile.
// Epilogue fuses per-channel sum / sum-of-squares (for batch-stat BN):
// lane pair (L, L+16) covers all 16 rows of channel n -> shfl_xor(16), then
// LDS accumulate across the 8 waves, one global atomic per channel per block.
// ---------------------------------------------------------------------------
__global__ void __launch_bounds__(256)
gemm_kernel(const _Float16* __restrict__ Xh, const _Float16* __restrict__ Bp,
            int Cpad, int Cout, float* __restrict__ Y,
            float* __restrict__ sums, float* __restrict__ sumsq) {
    __shared__ float ssum[16], ssq[16];
    const int tid = threadIdx.x;
    if (tid < 16) { ssum[tid] = 0.f; ssq[tid] = 0.f; }
    __syncthreads();

    const int wid  = tid >> 5;
    const int lane = tid & 31;
    const int mtile = blockIdx.x * 8 + wid;
    const int ntile = blockIdx.y;
    const int nchunks = Cpad >> 5;

    const int row = mtile * 16 + (lane & 15);
    const _Float16* arow = Xh + (size_t)row * Cpad + ((lane >= 16) ? 8 : 0);
    const _Float16* bptr = Bp + ((size_t)(ntile * nchunks) * 32 + lane) * 16;

    v8f acc = {};
    for (int c = 0; c < nchunks; ++c) {
        v8h alo = *(const v8h*)(arow + c * 32);
        v8h ahi = *(const v8h*)(arow + c * 32 + 16);
        v16h a;
#pragma unroll
        for (int e = 0; e < 8; ++e) { a[e] = alo[e]; a[8 + e] = ahi[e]; }
        v8h blo = *(const v8h*)(bptr + (size_t)c * 32 * 16);
        v8h bhi = *(const v8h*)(bptr + (size_t)c * 32 * 16 + 8);
        v16h bb;
#pragma unroll
        for (int e = 0; e < 8; ++e) { bb[e] = blo[e]; bb[8 + e] = bhi[e]; }
        acc = __builtin_amdgcn_wmma_f32_16x16x32_f16(false, a, false, bb,
                                                     (short)0, acc, false, false);
    }

    const int n = ntile * 16 + (lane & 15);
    const int mrow0 = mtile * 16 + ((lane >= 16) ? 8 : 0);
    float s = 0.f, s2 = 0.f;
#pragma unroll
    for (int r = 0; r < 8; ++r) {
        float v = acc[r];
        Y[(size_t)(mrow0 + r) * Cout + n] = v;
        s += v; s2 += v * v;
    }
    s  += __shfl_xor(s, 16, 32);
    s2 += __shfl_xor(s2, 16, 32);
    if (lane < 16) { atomicAdd(&ssum[lane], s); atomicAdd(&ssq[lane], s2); }
    __syncthreads();
    if (tid < 16) {
        atomicAdd(&sums[ntile * 16 + tid], ssum[tid]);
        atomicAdd(&sumsq[ntile * 16 + tid], ssq[tid]);
    }
}

// scale[c] = gamma * rsqrt(var + eps); shift[c] = beta - mean * scale
__global__ void bnfin_kernel(const float* __restrict__ sums, const float* __restrict__ sumsq,
                             const float* __restrict__ gamma, const float* __restrict__ beta,
                             float invM, int C,
                             float* __restrict__ scale, float* __restrict__ shift) {
    const int c = threadIdx.x;
    if (c < C) {
        float m = sums[c] * invM;
        float v = sumsq[c] * invM - m * m;
        float sc = gamma[c] * rsqrtf(v + 1e-5f);
        scale[c] = sc;
        shift[c] = beta[c] - m * sc;
    }
}

// BN + ReLU, emit padded f16 activations for the next WMMA layer.
__global__ void bnapply_kernel(const float* __restrict__ Y, const float* __restrict__ scale,
                               const float* __restrict__ shift, int C, int Cpad,
                               _Float16* __restrict__ Xh) {
    const int i = blockIdx.x * blockDim.x + threadIdx.x;  // total = M * Cpad
    const int c = i % Cpad;
    const int m = i / Cpad;
    float v = 0.f;
    if (c < C) {
        float y = Y[(size_t)m * C + c];
        v = fmaxf(y * scale[c] + shift[c], 0.f);
    }
    Xh[i] = (_Float16)v;
}

// BN + ReLU + max over K neighbors. transpose=1 writes (B,C,S) for d_out.
__global__ void maxpool_kernel(const float* __restrict__ Y, const float* __restrict__ scale,
                               const float* __restrict__ shift, int S, int K, int C,
                               float* __restrict__ out, int transpose) {
    const int i = blockIdx.x * blockDim.x + threadIdx.x;  // total = B*S*C
    const int c = i % C;
    const int q = i / C;
    const int b = q / S;
    const int s = q % S;
    const float sc = scale[c], sh = shift[c];
    const float* yb = Y + (size_t)q * K * C + c;
    float mv = -1e30f;
    for (int k = 0; k < K; ++k) {
        float v = fmaxf(yb[(size_t)k * C] * sc + sh, 0.f);
        mv = fmaxf(mv, v);
    }
    if (transpose) out[((size_t)b * C + c) * S + s] = mv;
    else           out[i] = mv;
}

// ---------------------------------------------------------------------------
// Host orchestration
// ---------------------------------------------------------------------------
namespace {
constexpr int B = 8;

// workspace byte offsets (all multiples of 256)
constexpr size_t OFF_FIDX   = 0;          //  32768 : int (B*1024)
constexpr size_t OFF_IDX    = 32768;      // 1048576: int (B*1024*32)
constexpr size_t OFF_NXYZ1  = 1081344;    //  98304 : f32 (B*1024*3)
constexpr size_t OFF_NXYZ2  = 1179648;    //  49152 : f32 (B*512*3)
constexpr size_t OFF_NXYZ3  = 1228800;    //  24576 : f32 (B*256*3)
constexpr size_t OFF_FEATS1 = 1253376;    // 1048576: f32 (B*1024*32)
constexpr size_t OFF_FEATS2 = 2301952;    // 1048576: f32 (B*512*64)
constexpr size_t OFF_STATS  = 3350528;    //   1024 : sums(128f) + sumsq(128f)
constexpr size_t OFF_SCALE  = 3351552;    //    512
constexpr size_t OFF_SHIFT  = 3352064;    //    512
constexpr size_t OFF_BPACK  = 3352576;    //  32768 : f16 packed weights
constexpr size_t OFF_XH     = 3385344;    // 16777216: f16 activations (M*Cpad)
constexpr size_t OFF_Y      = 20162560;   // 33554432: f32 conv output (M*Cout)

struct Ws {
    char* base;
    int*      fidx()   const { return (int*)(base + OFF_FIDX); }
    int*      idx()    const { return (int*)(base + OFF_IDX); }
    float*    nxyz1()  const { return (float*)(base + OFF_NXYZ1); }
    float*    nxyz2()  const { return (float*)(base + OFF_NXYZ2); }
    float*    nxyz3()  const { return (float*)(base + OFF_NXYZ3); }
    float*    feats1() const { return (float*)(base + OFF_FEATS1); }
    float*    feats2() const { return (float*)(base + OFF_FEATS2); }
    float*    sums()   const { return (float*)(base + OFF_STATS); }
    float*    sumsq()  const { return (float*)(base + OFF_STATS + 512); }
    float*    scale()  const { return (float*)(base + OFF_SCALE); }
    float*    shift()  const { return (float*)(base + OFF_SHIFT); }
    _Float16* bpack()  const { return (_Float16*)(base + OFF_BPACK); }
    _Float16* xh()     const { return (_Float16*)(base + OFF_XH); }
    float*    y()      const { return (float*)(base + OFF_Y); }
};

void run_layer(const Ws& ws, hipStream_t stream,
               const float* W, const float* gamma, const float* beta,
               int M, int Cin, int Cpad, int Cout,
               bool last, int S, int K, float* pooled_out, int transpose) {
    const int nchunks = Cpad / 32;
    const int ntiles  = Cout / 16;
    packw_kernel<<<ntiles * nchunks, 32, 0, stream>>>(W, Cin, Cout, nchunks, ws.bpack());
    hipMemsetAsync(ws.sums(), 0, 1024, stream);
    gemm_kernel<<<dim3(M / 128, ntiles), 256, 0, stream>>>(
        ws.xh(), ws.bpack(), Cpad, Cout, ws.y(), ws.sums(), ws.sumsq());
    bnfin_kernel<<<1, 128, 0, stream>>>(ws.sums(), ws.sumsq(), gamma, beta,
                                        1.0f / (float)M, Cout, ws.scale(), ws.shift());
    if (!last) {
        bnapply_kernel<<<(M * Cout) / 256, 256, 0, stream>>>(
            ws.y(), ws.scale(), ws.shift(), Cout, Cout, ws.xh());
    } else {
        maxpool_kernel<<<((M / K) * Cout) / 256, 256, 0, stream>>>(
            ws.y(), ws.scale(), ws.shift(), S, K, Cout, pooled_out, transpose);
    }
}
} // namespace

extern "C" void kernel_launch(void* const* d_in, const int* in_sizes, int n_in,
                              void* d_out, int out_size, void* d_ws, size_t ws_size,
                              hipStream_t stream) {
    (void)in_sizes; (void)n_in; (void)out_size; (void)ws_size;
    Ws ws{(char*)d_ws};
    const float* clouds = (const float*)d_in[0];
    // params: 3 modules x layers x (W, gamma, beta), flattened depth-first
    const float* P[21];
    for (int i = 0; i < 21; ++i) P[i] = (const float*)d_in[1 + i];
    float* out = (float*)d_out;

    // ---- Module 1: N=16384, S=1024, K=32, r=0.1; Cin=3 (pad 32); 3->32->32
    {
        const int N = 16384, S = 1024, K = 32, M = B * S * K;
        fps_kernel<16><<<B, 1024, 0, stream>>>(clouds, N, S, ws.fidx());
        ballquery_kernel<<<(B * S) / 8, 256, 0, stream>>>(
            clouds, ws.fidx(), N, S, K, 0.01f, ws.idx(), ws.nxyz1());
        group_kernel<<<M / 256, 256, 0, stream>>>(
            clouds, nullptr, ws.idx(), ws.nxyz1(), N, S, K, 0, 32, ws.xh());
        run_layer(ws, stream, P[0], P[1], P[2], M, 3, 32, 32, false, S, K, nullptr, 0);
        run_layer(ws, stream, P[3], P[4], P[5], M, 32, 32, 32, true, S, K, ws.feats1(), 0);
    }
    // ---- Module 2: N=1024, S=512, K=32, r=0.2; Cin=35 (pad 64); 35->64->64
    {
        const int N = 1024, S = 512, K = 32, M = B * S * K;
        fps_kernel<1><<<B, 1024, 0, stream>>>(ws.nxyz1(), N, S, ws.fidx());
        ballquery_kernel<<<(B * S) / 8, 256, 0, stream>>>(
            ws.nxyz1(), ws.fidx(), N, S, K, 0.04f, ws.idx(), ws.nxyz2());
        group_kernel<<<M / 256, 256, 0, stream>>>(
            ws.nxyz1(), ws.feats1(), ws.idx(), ws.nxyz2(), N, S, K, 32, 64, ws.xh());
        run_layer(ws, stream, P[6], P[7], P[8], M, 35, 64, 64, false, S, K, nullptr, 0);
        run_layer(ws, stream, P[9], P[10], P[11], M, 64, 64, 64, true, S, K, ws.feats2(), 0);
    }
    // ---- Module 3: N=512, S=256, K=16, r=0.4; Cin=67 (pad 96); 67->64->64->128
    {
        const int N = 512, S = 256, K = 16, M = B * S * K;
        fps_kernel<1><<<B, 1024, 0, stream>>>(ws.nxyz2(), N, S, ws.fidx());
        ballquery_kernel<<<(B * S) / 8, 256, 0, stream>>>(
            ws.nxyz2(), ws.fidx(), N, S, K, 0.16f, ws.idx(), ws.nxyz3());
        group_kernel<<<M / 256, 256, 0, stream>>>(
            ws.nxyz2(), ws.feats2(), ws.idx(), ws.nxyz3(), N, S, K, 64, 96, ws.xh());
        run_layer(ws, stream, P[12], P[13], P[14], M, 67, 96, 64, false, S, K, nullptr, 0);
        run_layer(ws, stream, P[15], P[16], P[17], M, 64, 64, 64, false, S, K, nullptr, 0);
        run_layer(ws, stream, P[18], P[19], P[20], M, 64, 64, 128, true, S, K, out, 1);
    }
}
